// DecomposeWHAR_32143535243597
// MI455X (gfx1250) — matrix operand order repaired and checked by reading, not compiled, not run on hardware
//
#include <hip/hip_runtime.h>
#include <hip/hip_bf16.h>
#include <math.h>

typedef __attribute__((ext_vector_type(16))) _Float16 v16h;
typedef __attribute__((ext_vector_type(8)))  _Float16 v8h;
typedef __attribute__((ext_vector_type(8)))  float    v8f;

// ---------------- model dimensions ----------------
constexpr int B0v = 128, NSv = 5, Lv = 128, Mv = 9, Dv = 64, Pv = 8, Sv = 4, Kdw = 5;
constexpr int Tn   = Lv / Sv;        // 32
constexpr int Bful = B0v * NSv;      // 640
constexpr int MD   = Mv * Dv;        // 576
constexpr int DMm  = NSv * Dv;       // 320
constexpr int DIm  = 2 * DMm;        // 640
constexpr int DRv  = 20, DSv = 16;
constexpr int Hh   = 8;
constexpr int DATT = Dv * Tn;        // 2048
constexpr int DHh  = DATT / Hh;      // 256
constexpr int NCv  = 17;

constexpr int XB_N   = Bful * Mv * Dv * Tn;   // 11,796,480
constexpr int FFN2_N = Bful * Dv * Tn;        // 1,310,720
constexpr int FEAT_N = B0v * Tn * DMm;        // 1,310,720
constexpr int CONV_N = B0v * Tn * DIm;        // 2,621,440
constexpr int SCAN_N = B0v * DIm;             // 81,920
constexpr int XA_N   = Bful * DATT;           // 1,310,720
constexpr int ATTN_N = B0v * Hh * NSv;        // 5,120

__device__ __forceinline__ float gelu_tanh(float x) {
    float x3 = x * x * x;
    return 0.5f * x * (1.f + tanhf(0.7978845608028654f * (x + 0.044715f * x3)));
}
__device__ __forceinline__ float siluf(float x) { return x / (1.f + expf(-x)); }
__device__ __forceinline__ float softplusf(float x) { return (x > 20.f) ? x : log1pf(expf(x)); }

// Build v16h fragment from two contiguous 8-half LDS spans.
__device__ __forceinline__ v16h frag16(const _Float16* p0, const _Float16* p1) {
    v8h lo = *(const v8h*)p0;
    v8h hi = *(const v8h*)p1;
    return __builtin_shufflevector(lo, hi, 0, 1, 2, 3, 4, 5, 6, 7,
                                   8, 9, 10, 11, 12, 13, 14, 15);
}

// ---------------- fast WMMA GEMM (M%64==0, N%64==0, K%32==0) ----------------
// C[M x N] = A[M x K] * B[K x N] (+bias[n]); fp32 in/out, f16 WMMA compute.
// 256 threads = 8 waves (4x2); block tile 64x64; each wave: 16x32 (2 WMMAs/K-step).
__global__ void k_gemm_wmma_fast(const float* __restrict__ A, int lda,
                                 const float* __restrict__ Bm, int ldb,
                                 float* __restrict__ C, int ldc,
                                 const float* __restrict__ bias, int Kd) {
    __shared__ __align__(16) _Float16 As[64][40];   // row-major A tile (64 x 32)
    __shared__ __align__(16) _Float16 BsT[64][40];  // transposed B tile: [n][k]
    const int tid  = threadIdx.x;
    const int lane = tid & 31;
    const int wave = tid >> 5;       // 0..7
    const int wr   = wave >> 1;      // 0..3 : 16-row strip
    const int wc   = wave & 1;       // 0..1 : 32-col strip
    const int tile_m = blockIdx.y * 64;
    const int tile_n = blockIdx.x * 64;

    // A tile: 64x32 floats, 8 consecutive floats per thread (global_load_b128 x2)
    const int ar = tid >> 2;         // 0..63
    const int ac = (tid & 3) * 8;    // 0,8,16,24
    // B tile: 32x64 floats, 8 consecutive floats per thread, stored transposed
    const int brr = tid >> 3;        // 0..31
    const int bcc = (tid & 7) * 8;   // 0..56

    const int mrow = wr * 16 + (lane & 15);
    const bool hi  = lane >= 16;
    const int abase = hi ? 16 : 0;   // A halves: k = abase..abase+15 (contiguous)
    const int kb    = hi ? 16 : 0;   // B halves: k = kb..kb+15 (contiguous in BsT row)
    const int n0    = wc * 32 + (lane & 15);

    v8f acc0 = {}, acc1 = {};
    for (int k0 = 0; k0 < Kd; k0 += 32) {
        const float* ap = A + (size_t)(tile_m + ar) * lda + k0 + ac;
        float4 a0 = *(const float4*)ap;
        float4 a1 = *(const float4*)(ap + 4);
        v8h ah = { (_Float16)a0.x, (_Float16)a0.y, (_Float16)a0.z, (_Float16)a0.w,
                   (_Float16)a1.x, (_Float16)a1.y, (_Float16)a1.z, (_Float16)a1.w };
        *(v8h*)&As[ar][ac] = ah;

        const float* bp = Bm + (size_t)(k0 + brr) * ldb + tile_n + bcc;
        float4 b0 = *(const float4*)bp;
        float4 b1 = *(const float4*)(bp + 4);
        BsT[bcc + 0][brr] = (_Float16)b0.x;
        BsT[bcc + 1][brr] = (_Float16)b0.y;
        BsT[bcc + 2][brr] = (_Float16)b0.z;
        BsT[bcc + 3][brr] = (_Float16)b0.w;
        BsT[bcc + 4][brr] = (_Float16)b1.x;
        BsT[bcc + 5][brr] = (_Float16)b1.y;
        BsT[bcc + 6][brr] = (_Float16)b1.z;
        BsT[bcc + 7][brr] = (_Float16)b1.w;

        if (k0 + 32 < Kd) {  // prefetch next K tile into cache
            __builtin_prefetch(ap + 32, 0, 0);
            __builtin_prefetch(bp + (size_t)32 * ldb, 0, 0);
        }
        __syncthreads();

        v16h af  = frag16(&As[mrow][abase], &As[mrow][abase + 8]);
        v16h bf0 = frag16(&BsT[n0][kb], &BsT[n0][kb + 8]);
        v16h bf1 = frag16(&BsT[n0 + 16][kb], &BsT[n0 + 16][kb + 8]);
        acc0 = __builtin_amdgcn_wmma_f32_16x16x32_f16(false, af, false, bf0,
                                                      (short)0, acc0, false, false);
        acc1 = __builtin_amdgcn_wmma_f32_16x16x32_f16(false, af, false, bf1,
                                                      (short)0, acc1, false, false);
        __syncthreads();
    }

    const int ng = tile_n + n0;
    const int mb = tile_m + wr * 16 + ((lane >> 4) << 3);
    const float bv0 = bias ? bias[ng] : 0.f;
    const float bv1 = bias ? bias[ng + 16] : 0.f;
#pragma unroll
    for (int i = 0; i < 8; ++i) {
        C[(size_t)(mb + i) * ldc + ng]      = acc0[i] + bv0;
        C[(size_t)(mb + i) * ldc + ng + 16] = acc1[i] + bv1;
    }
}

// ---------------- generic WMMA GEMM (any M/N/K, optional softplus) ----------------
// 128 threads = 4 waves (2x2); block tile 32x32. Clamped unconditional loads.
__global__ void k_gemm_wmma_gen(const float* __restrict__ A, int lda,
                                const float* __restrict__ Bm, int ldb,
                                float* __restrict__ C, int ldc,
                                const float* __restrict__ bias,
                                int Mr, int Nc, int Kd, int act) {
    __shared__ __align__(16) _Float16 As[32][40];
    __shared__ __align__(16) _Float16 BsT[32][40];  // [n][k]
    const int tid  = threadIdx.x;
    const int wave = tid >> 5;
    const int lane = tid & 31;
    const int wr = wave >> 1, wc = wave & 1;
    const int tile_m = blockIdx.y * 32;
    const int tile_n = blockIdx.x * 32;

    const int mrow  = wr * 16 + (lane & 15);
    const bool hi   = lane >= 16;
    const int abase = hi ? 16 : 0;
    const int kb    = hi ? 16 : 0;
    const int ncol  = wc * 16 + (lane & 15);

    v8f acc = {};
    for (int k0 = 0; k0 < Kd; k0 += 32) {
        for (int i = tid; i < 32 * 32; i += 128) {
            int r = i >> 5, c = i & 31;
            // clamped unconditional loads -> no EXEC branching, vectorizable waits
            int gr = tile_m + r, gc = k0 + c;
            float av = A[(size_t)min(gr, Mr - 1) * lda + min(gc, Kd - 1)];
            As[r][c] = (_Float16)((gr < Mr && gc < Kd) ? av : 0.f);
            int br = k0 + r, bc = tile_n + c;
            float bvv = Bm[(size_t)min(br, Kd - 1) * ldb + min(bc, Nc - 1)];
            BsT[c][r] = (_Float16)((br < Kd && bc < Nc) ? bvv : 0.f);
        }
        __syncthreads();
        v16h af = frag16(&As[mrow][abase], &As[mrow][abase + 8]);
        v16h bf = frag16(&BsT[ncol][kb], &BsT[ncol][kb + 8]);
        acc = __builtin_amdgcn_wmma_f32_16x16x32_f16(false, af, false, bf,
                                                     (short)0, acc, false, false);
        __syncthreads();
    }

    const int n     = tile_n + ncol;
    const int mbase = tile_m + wr * 16 + ((lane >> 4) << 3);
    const float bv  = (bias != nullptr && n < Nc) ? bias[n] : 0.f;
#pragma unroll
    for (int i = 0; i < 8; ++i) {
        int m = mbase + i;
        if (m < Mr && n < Nc) {
            float r = acc[i] + bv;
            if (act == 1) r = softplusf(r);
            C[(size_t)m * ldc + n] = r;
        }
    }
}

// ---------------- elementwise / small kernels ----------------

__global__ void k_embed(const float* __restrict__ inp, const float* __restrict__ ew,
                        const float* __restrict__ eb, float* __restrict__ xb) {
    int idx = blockIdx.x * blockDim.x + threadIdx.x;
    if (idx >= XB_N) return;
    int t = idx % Tn, d = (idx / Tn) % Dv, m = (idx / (Tn * Dv)) % Mv, b = idx / (Tn * Dv * Mv);
    float acc = eb[d];
#pragma unroll
    for (int p = 0; p < Pv; ++p) {
        int l = t * Sv - 2 + p;
        if (l >= 0 && l < Lv) acc += ew[d * Pv + p] * inp[(b * Lv + l) * Mv + m];
    }
    xb[idx] = acc;
}

__global__ void k_dwconv(const float* __restrict__ xb, const float* __restrict__ w,
                         const float* __restrict__ cb, const float* __restrict__ g,
                         const float* __restrict__ bb, const float* __restrict__ mm,
                         const float* __restrict__ vv, float* __restrict__ u, int l) {
    int idx = blockIdx.x * blockDim.x + threadIdx.x;
    if (idx >= XB_N) return;
    int t = idx % Tn, c = (idx / Tn) % MD, b = idx / (Tn * MD);
    const float* xr = xb + (size_t)(b * MD + c) * Tn;
    int lc = l * MD + c;
    float acc = cb[lc];
#pragma unroll
    for (int j = 0; j < Kdw; ++j) {
        int tt = t - 2 + j;
        if (tt >= 0 && tt < Tn) acc += w[lc * Kdw + j] * xr[tt];
    }
    float bn = (acc - mm[lc]) * rsqrtf(vv[lc] + 1e-5f) * g[lc] + bb[lc];
    u[idx] = xr[t] + bn;
}

__global__ void k_ffn1a(const float* __restrict__ u, const float* __restrict__ w1,
                        const float* __restrict__ b1, float* __restrict__ h, int l) {
    int idx = blockIdx.x * blockDim.x + threadIdx.x;
    if (idx >= XB_N) return;
    int t = idx % Tn, o = (idx / Tn) % Dv, m = (idx / (Tn * Dv)) % Mv, b = idx / (Tn * Dv * Mv);
    const float* ur = u + (size_t)((b * Mv + m) * Dv) * Tn + t;
    const float* wr = w1 + (size_t)((l * Mv + m) * Dv + o) * Dv;
    float acc = b1[(l * Mv + m) * Dv + o];
#pragma unroll 8
    for (int d = 0; d < Dv; ++d) acc += ur[d * Tn] * wr[d];
    h[idx] = gelu_tanh(acc);
}

__global__ void k_ffn1b(const float* __restrict__ h, const float* __restrict__ w2,
                        const float* __restrict__ b2, float* __restrict__ u, int l) {
    int idx = blockIdx.x * blockDim.x + threadIdx.x;
    if (idx >= XB_N) return;
    int t = idx % Tn, d = (idx / Tn) % Dv, m = (idx / (Tn * Dv)) % Mv, b = idx / (Tn * Dv * Mv);
    const float* hr = h + (size_t)((b * Mv + m) * Dv) * Tn + t;
    const float* wr = w2 + (size_t)((l * Mv + m) * Dv + d) * Dv;
    float acc = b2[(l * Mv + m) * Dv + d];
#pragma unroll 8
    for (int o = 0; o < Dv; ++o) acc += hr[o * Tn] * wr[o];
    u[idx] += acc;
}

__global__ void k_ffn2(const float* __restrict__ u, const float* __restrict__ w1,
                       const float* __restrict__ b1, const float* __restrict__ w2,
                       const float* __restrict__ b2, float* __restrict__ xb, int l) {
    int idx = blockIdx.x * blockDim.x + threadIdx.x;
    if (idx >= FFN2_N) return;
    int t = idx % Tn, d = (idx / Tn) % Dv, b = idx / (Tn * Dv);
    float v[Mv], h2[Mv];
#pragma unroll
    for (int m = 0; m < Mv; ++m) v[m] = u[(size_t)((b * Mv + m) * Dv + d) * Tn + t];
#pragma unroll
    for (int o = 0; o < Mv; ++o) {
        float a = b1[(l * Dv + d) * Mv + o];
#pragma unroll
        for (int m = 0; m < Mv; ++m) a += v[m] * w1[((l * Dv + d) * Mv + o) * Mv + m];
        h2[o] = gelu_tanh(a);
    }
#pragma unroll
    for (int m = 0; m < Mv; ++m) {
        float a = b2[(l * Dv + d) * Mv + m];
#pragma unroll
        for (int o = 0; o < Mv; ++o) a += h2[o] * w2[((l * Dv + d) * Mv + m) * Mv + o];
        xb[(size_t)((b * Mv + m) * Dv + d) * Tn + t] = v[m] + a;
    }
}

__global__ void k_feat(const float* __restrict__ xb, float* __restrict__ feat) {
    int idx = blockIdx.x * blockDim.x + threadIdx.x;
    if (idx >= FEAT_N) return;
    int c = idx % DMm, t = (idx / DMm) % Tn, b0 = idx / (DMm * Tn);
    int s = c / Dv, d = c % Dv;
    float acc = 0.f;
#pragma unroll
    for (int m = 0; m < Mv; ++m) acc += xb[(size_t)(((b0 * NSv + s) * Mv + m) * Dv + d) * Tn + t];
    feat[idx] = acc * (1.f / 9.f);
}

__global__ void k_convsilu(const float* __restrict__ xz, const float* __restrict__ w,
                           const float* __restrict__ cb, float* __restrict__ xs) {
    int idx = blockIdx.x * blockDim.x + threadIdx.x;
    if (idx >= CONV_N) return;
    int c = idx % DIm, t = (idx / DIm) % Tn, b0 = idx / (DIm * Tn);
    float acc = cb[c];
#pragma unroll
    for (int j = 0; j < 4; ++j) {
        int tt = t - 3 + j;
        if (tt >= 0) acc += w[c * 4 + j] * xz[(size_t)(b0 * Tn + tt) * (2 * DIm) + c];
    }
    xs[idx] = siluf(acc);
}

__global__ void k_scan(const float* __restrict__ dt, const float* __restrict__ xs,
                       const float* __restrict__ dbc, const float* __restrict__ xz,
                       const float* __restrict__ alog, const float* __restrict__ mD,
                       float* __restrict__ y) {
    int idx = blockIdx.x * blockDim.x + threadIdx.x;
    if (idx >= SCAN_N) return;
    int c = idx % DIm, b0 = idx / DIm;
    float Ac[DSv], h[DSv];
#pragma unroll
    for (int s = 0; s < DSv; ++s) { Ac[s] = -expf(alog[c * DSv + s]); h[s] = 0.f; }
    float Dc = mD[c];
    for (int t = 0; t < Tn; ++t) {
        int row = b0 * Tn + t;
        float dtv = dt[(size_t)row * DIm + c];
        float xv  = xs[(size_t)row * DIm + c];
        float acc = Dc * xv;
        const float* dr = dbc + (size_t)row * (DRv + 2 * DSv);
#pragma unroll
        for (int s = 0; s < DSv; ++s) {
            float dA = expf(dtv * Ac[s]);
            h[s] = dA * h[s] + dtv * dr[DRv + s] * xv;
            acc += h[s] * dr[DRv + DSv + s];
        }
        float z = xz[(size_t)row * (2 * DIm) + DIm + c];
        y[(size_t)row * DIm + c] = acc * siluf(z);
    }
}

__global__ void k_ln1(float* __restrict__ feat, const float* __restrict__ yout,
                      const float* __restrict__ g, const float* __restrict__ bb) {
    int r = blockIdx.x * blockDim.x + threadIdx.x;
    if (r >= B0v * Tn) return;
    float* fr = feat + (size_t)r * DMm;
    const float* yr = yout + (size_t)r * DMm;
    float mu = 0.f;
    for (int i = 0; i < DMm; ++i) mu += fr[i] + yr[i];
    mu /= (float)DMm;
    float var = 0.f;
    for (int i = 0; i < DMm; ++i) { float d = fr[i] + yr[i] - mu; var += d * d; }
    var /= (float)DMm;
    float inv = rsqrtf(var + 1e-5f);
    for (int i = 0; i < DMm; ++i) {
        float x = fr[i] + yr[i];
        fr[i] = (x - mu) * inv * g[i] + bb[i];
    }
}

__global__ void k_xa(const float* __restrict__ feat, float* __restrict__ xa) {
    int idx = blockIdx.x * blockDim.x + threadIdx.x;
    if (idx >= XA_N) return;
    int i = idx % DATT, s = (idx / DATT) % NSv, b0 = idx / (DATT * NSv);
    int d = i / Tn, t = i % Tn;
    xa[idx] = feat[(size_t)(b0 * Tn + t) * DMm + s * Dv + d];
}

__global__ void k_attn(const float* __restrict__ q, const float* __restrict__ k,
                       const float* __restrict__ v, float* __restrict__ o) {
    int idx = blockIdx.x * blockDim.x + threadIdx.x;
    if (idx >= ATTN_N) return;
    int lq = idx % NSv, hh = (idx / NSv) % Hh, b0 = idx / (NSv * Hh);
    const float* qr = q + (size_t)(b0 * NSv + lq) * DATT + hh * DHh;
    float sc[NSv], mx = -1e30f;
#pragma unroll
    for (int s = 0; s < NSv; ++s) {
        const float* kr = k + (size_t)(b0 * NSv + s) * DATT + hh * DHh;
        float a = 0.f;
        for (int e = 0; e < DHh; ++e) a += qr[e] * kr[e];
        a *= 0.0625f; // 1/sqrt(256)
        sc[s] = a;
        mx = fmaxf(mx, a);
    }
    float den = 0.f;
#pragma unroll
    for (int s = 0; s < NSv; ++s) { sc[s] = expf(sc[s] - mx); den += sc[s]; }
    float rden = 1.f / den;
#pragma unroll
    for (int s = 0; s < NSv; ++s) sc[s] *= rden;
    float* orr = o + (size_t)(b0 * NSv + lq) * DATT + hh * DHh;
    for (int e = 0; e < DHh; ++e) {
        float a = 0.f;
#pragma unroll
        for (int s = 0; s < NSv; ++s) a += sc[s] * v[(size_t)(b0 * NSv + s) * DATT + hh * DHh + e];
        orr[e] = a;
    }
}

__global__ void k_ln2(const float* __restrict__ xa, const float* __restrict__ op,
                      const float* __restrict__ g, const float* __restrict__ bb,
                      float* __restrict__ out) {
    __shared__ float tv[DATT];
    __shared__ float red[256];
    int r = blockIdx.x, tid = threadIdx.x;
    float ps = 0.f;
    for (int i = tid; i < DATT; i += 256) {
        float x = xa[(size_t)r * DATT + i] + op[(size_t)r * DATT + i];
        tv[i] = x;
        ps += x;
    }
    red[tid] = ps;
    __syncthreads();
    for (int s = 128; s > 0; s >>= 1) { if (tid < s) red[tid] += red[tid + s]; __syncthreads(); }
    float mu = red[0] / (float)DATT;
    __syncthreads();
    float pv = 0.f;
    for (int i = tid; i < DATT; i += 256) { float d = tv[i] - mu; pv += d * d; }
    red[tid] = pv;
    __syncthreads();
    for (int s = 128; s > 0; s >>= 1) { if (tid < s) red[tid] += red[tid + s]; __syncthreads(); }
    float inv = rsqrtf(red[0] / (float)DATT + 1e-5f);
    for (int i = tid; i < DATT; i += 256)
        out[(size_t)r * DATT + i] = (tv[i] - mu) * inv * g[i] + bb[i];
}

// ---------------- launcher ----------------
extern "C" void kernel_launch(void* const* d_in, const int* in_sizes, int n_in,
                              void* d_out, int out_size, void* d_ws, size_t ws_size,
                              hipStream_t stream) {
    (void)in_sizes; (void)n_in; (void)out_size; (void)ws_size;
    const float* inp     = (const float*)d_in[0];
    const float* emb_w   = (const float*)d_in[1];
    const float* emb_b   = (const float*)d_in[2];
    const float* dw_w    = (const float*)d_in[3];
    const float* dw_b    = (const float*)d_in[4];
    const float* bn_g    = (const float*)d_in[5];
    const float* bn_b    = (const float*)d_in[6];
    const float* bn_m    = (const float*)d_in[7];
    const float* bn_v    = (const float*)d_in[8];
    const float* f1w1    = (const float*)d_in[9];
    const float* f1b1    = (const float*)d_in[10];
    const float* f1w2    = (const float*)d_in[11];
    const float* f1b2    = (const float*)d_in[12];
    const float* f2w1    = (const float*)d_in[13];
    const float* f2b1    = (const float*)d_in[14];
    const float* f2w2    = (const float*)d_in[15];
    const float* f2b2    = (const float*)d_in[16];
    const float* m_in    = (const float*)d_in[17];
    const float* m_convw = (const float*)d_in[18];
    const float* m_convb = (const float*)d_in[19];
    const float* m_xproj = (const float*)d_in[20];
    const float* m_dtw   = (const float*)d_in[21];
    const float* m_dtb   = (const float*)d_in[22];
    const float* m_Alog  = (const float*)d_in[23];
    const float* m_D     = (const float*)d_in[24];
    const float* m_out   = (const float*)d_in[25];
    const float* ln1_g   = (const float*)d_in[26];
    const float* ln1_b   = (const float*)d_in[27];
    const float* wq      = (const float*)d_in[28];
    const float* bq      = (const float*)d_in[29];
    const float* wk      = (const float*)d_in[30];
    const float* bk      = (const float*)d_in[31];
    const float* wv      = (const float*)d_in[32];
    const float* bv      = (const float*)d_in[33];
    const float* wo      = (const float*)d_in[34];
    const float* bo      = (const float*)d_in[35];
    const float* ln2_g   = (const float*)d_in[36];
    const float* ln2_b   = (const float*)d_in[37];
    const float* fc_w    = (const float*)d_in[38];
    const float* fc_b    = (const float*)d_in[39];
    float* out = (float*)d_out;
    float* ws  = (float*)d_ws;

    // workspace layout (floats), lifetime-based aliasing:
    float* XB = ws;                  // conv-stage state; reused for attention bufs later
    float* U  = ws + XB_N;           // layer temp; reused for mamba bufs later
    float* H1 = ws + 2 * XB_N;       // layer temp
    // mamba region (aliases U/H1, live only after conv layers are done):
    float* FEAT = U;
    float* XZ   = FEAT + FEAT_N;               // 4096 x 1280
    float* XS   = XZ + B0v * Tn * 2 * DIm;     // 4096 x 640
    float* DBC  = XS + CONV_N;                 // 4096 x 52
    float* DT   = DBC + B0v * Tn * (DRv + 2 * DSv);
    float* Yb   = DT + CONV_N;                 // 4096 x 640
    float* YOUT = Yb + CONV_N;                 // 4096 x 320
    // attention region (aliases XB, live only after k_feat consumed XB):
    float* XA = XB;
    float* Qb = XA + XA_N;
    float* Kb = Qb + XA_N;
    float* Vb = Kb + XA_N;
    float* Ob = Vb + XA_N;
    float* OP = Ob + XA_N;

    auto g256 = [](int n) { return dim3((unsigned)((n + 255) / 256)); };
    const dim3 blk256(256), blkGen(128), blkFast(256);

    k_embed<<<g256(XB_N), blk256, 0, stream>>>(inp, emb_w, emb_b, XB);

    for (int l = 0; l < 2; ++l) {
        k_dwconv<<<g256(XB_N), blk256, 0, stream>>>(XB, dw_w, dw_b, bn_g, bn_b, bn_m, bn_v, U, l);
        k_ffn1a<<<g256(XB_N), blk256, 0, stream>>>(U, f1w1, f1b1, H1, l);
        k_ffn1b<<<g256(XB_N), blk256, 0, stream>>>(H1, f1w2, f1b2, U, l);
        k_ffn2<<<g256(FFN2_N), blk256, 0, stream>>>(U, f2w1, f2b1, f2w2, f2b2, XB, l);
    }

    k_feat<<<g256(FEAT_N), blk256, 0, stream>>>(XB, FEAT);

    // xz = feat @ m_in             (4096 x 1280, K=320)  -> fast path
    k_gemm_wmma_fast<<<dim3((2 * DIm) / 64, (B0v * Tn) / 64), blkFast, 0, stream>>>(
        FEAT, DMm, m_in, 2 * DIm, XZ, 2 * DIm, nullptr, DMm);
    k_convsilu<<<g256(CONV_N), blk256, 0, stream>>>(XZ, m_convw, m_convb, XS);
    // dbc = xs @ m_xproj           (4096 x 52, K=640)    -> generic (N ragged)
    k_gemm_wmma_gen<<<dim3(2, (B0v * Tn) / 32), blkGen, 0, stream>>>(
        XS, DIm, m_xproj, DRv + 2 * DSv, DBC, DRv + 2 * DSv, nullptr,
        B0v * Tn, DRv + 2 * DSv, DIm, 0);
    // dt = softplus(dbc[:, :20] @ m_dtw + m_dtb)  (4096 x 640, K=20) -> generic (K ragged)
    k_gemm_wmma_gen<<<dim3(DIm / 32, (B0v * Tn) / 32), blkGen, 0, stream>>>(
        DBC, DRv + 2 * DSv, m_dtw, DIm, DT, DIm, m_dtb, B0v * Tn, DIm, DRv, 1);
    k_scan<<<g256(SCAN_N), blk256, 0, stream>>>(DT, XS, DBC, XZ, m_Alog, m_D, Yb);
    // yout = y @ m_out             (4096 x 320, K=640)   -> fast path
    k_gemm_wmma_fast<<<dim3(DMm / 64, (B0v * Tn) / 64), blkFast, 0, stream>>>(
        Yb, DIm, m_out, DMm, YOUT, DMm, nullptr, DIm);
    k_ln1<<<g256(B0v * Tn), blk256, 0, stream>>>(FEAT, YOUT, ln1_g, ln1_b);

    k_xa<<<g256(XA_N), blk256, 0, stream>>>(FEAT, XA);
    // Q/K/V projections            (640 x 2048, K=2048)  -> fast path
    k_gemm_wmma_fast<<<dim3(DATT / 64, Bful / 64), blkFast, 0, stream>>>(
        XA, DATT, wq, DATT, Qb, DATT, bq, DATT);
    k_gemm_wmma_fast<<<dim3(DATT / 64, Bful / 64), blkFast, 0, stream>>>(
        XA, DATT, wk, DATT, Kb, DATT, bk, DATT);
    k_gemm_wmma_fast<<<dim3(DATT / 64, Bful / 64), blkFast, 0, stream>>>(
        XA, DATT, wv, DATT, Vb, DATT, bv, DATT);
    k_attn<<<g256(ATTN_N), blk256, 0, stream>>>(Qb, Kb, Vb, Ob);
    // output projection            (640 x 2048, K=2048)  -> fast path
    k_gemm_wmma_fast<<<dim3(DATT / 64, Bful / 64), blkFast, 0, stream>>>(
        Ob, DATT, wo, DATT, OP, DATT, bo, DATT);
    // ln2 + write xflat to d_out
    k_ln2<<<dim3(Bful), blk256, 0, stream>>>(XA, OP, ln2_g, ln2_b, out);
    // pred = xflat @ fc_w + fc_b   (128 x 17, K=10240)   -> generic (N ragged)
    k_gemm_wmma_gen<<<dim3((NCv + 31) / 32, B0v / 32), blkGen, 0, stream>>>(
        out, NSv * DATT, fc_w, NCv, out + (size_t)B0v * NSv * DATT, NCv, fc_b,
        B0v, NCv, NSv * DATT, 0);
}
